// ThreeWLGNNNet_80633716015167
// MI455X (gfx1250) — matrix operand order repaired and checked by reading, not compiled, not run on hardware
//
#include <hip/hip_runtime.h>
#include <hip/hip_bf16.h>

// 3WL-GNN forward for MI455X (gfx1250), wave32 + WMMA bf16.
// Layout-driven design: activations channel-last so every conv/readout GEMM
// loads A-fragments straight from global; per-channel matmuls use a
// transposed m2 copy so they need no LDS at all (L2-resident, 192MB).

typedef __attribute__((ext_vector_type(16))) __bf16 v16bf;
typedef __attribute__((ext_vector_type(8)))  __bf16 v8bf;
typedef __attribute__((ext_vector_type(8)))  float  v8f;

#define NDIM  384
#define NNSQ  (384 * 384)
#define EEDGE 49152
#define CTOT  257            // 17 + 3*80

__device__ __forceinline__ float  bf2f(__bf16 x) { return (float)x; }
__device__ __forceinline__ __bf16 f2bf(float f)  { return (__bf16)f; }

__device__ __forceinline__ v16bf mk16(v8bf lo, v8bf hi) {
  v16bf r;
#pragma unroll
  for (int i = 0; i < 8; ++i) { r[i] = lo[i]; r[8 + i] = hi[i]; }
  return r;
}

// ---------------------------------------------------------------------------
// Unified GEMM: Y = act(X[M][Kpad] * W[O][Kreal]^T + b)
// X rows are K-contiguous -> A-fragments load directly from global.
// W tile (16 outs x Kpad) converted fp32->bf16 into LDS once per block.
// Block = 128 threads (4 waves): 64 rows x 16 outs per block.
// mode 0: Y[row*ldy + o]                  (channel-last activations)
// mode 1: Y[o*NNSQ + row]                 (channel-first, contiguous b128)
// mode 2: Y[o*NNSQ + (row%384)*384 + row/384]  (channel-first transposed)
// ---------------------------------------------------------------------------
__global__ __launch_bounds__(128)
void gemm_xwT_wmma(const __bf16* __restrict__ X, const float* __restrict__ W,
                   const float* __restrict__ Bv, __bf16* __restrict__ Y,
                   int Kpad, int Kreal, int O, int ldy, int relu, int mode) {
  __shared__ __align__(16) __bf16 Wt[16 * (544 + 8)];
  const int tid  = threadIdx.x;
  const int wave = tid >> 5, lane = tid & 31;
  const int half = lane >> 4, lr = lane & 15;
  const int o0 = blockIdx.y * 16;
  const int e0 = blockIdx.x * 64 + wave * 16;
  const int SW = Kpad + 8;

  for (int idx = tid; idx < 16 * Kpad; idx += 128) {
    int r = idx / Kpad, k = idx - r * Kpad;
    int o = o0 + r;
    float v = (o < O && k < Kreal) ? W[o * Kreal + k] : 0.f;
    Wt[r * SW + k] = f2bf(v);
  }
  __syncthreads();

  v8f acc = {};
  const __bf16* Xrow = X + (size_t)(e0 + lr) * Kpad;
  for (int kk = 0; kk < Kpad; kk += 32) {
    v16bf a = mk16(*(const v8bf*)(Xrow + kk + (half ? 16 : 0)),
                   *(const v8bf*)(Xrow + kk + 8 + (half ? 16 : 0)));
    const __bf16* br = &Wt[lr * SW + kk + (half ? 16 : 0)];
    v16bf b = mk16(*(const v8bf*)br, *(const v8bf*)(br + 8));
    acc = __builtin_amdgcn_wmma_f32_16x16x32_bf16(false, a, false, b,
                                                  (short)0, acc, false, false);
  }

  const int o = o0 + lr;
  if (o < O) {
    float bias = Bv[o];
#pragma unroll
    for (int r = 0; r < 8; ++r) {
      float v = acc[r] + bias;
      if (relu) v = fmaxf(v, 0.f);
      acc[r] = v;
    }
    if (mode == 0) {
#pragma unroll
      for (int r = 0; r < 8; ++r)
        Y[(size_t)(e0 + half * 8 + r) * ldy + o] = f2bf(acc[r]);
    } else if (mode == 1) {
      v8bf pk;
#pragma unroll
      for (int r = 0; r < 8; ++r) pk[r] = f2bf(acc[r]);
      *(v8bf*)(Y + (size_t)o * NNSQ + e0 + half * 8) = pk;
    } else {
#pragma unroll
      for (int r = 0; r < 8; ++r) {
        int p = e0 + half * 8 + r;
        Y[(size_t)o * NNSQ + (p % NDIM) * NDIM + p / NDIM] = f2bf(acc[r]);
      }
    }
  }
}

// ---------------------------------------------------------------------------
// Per-channel matmul, LDS-free: mult[c] = m1[c] @ m2[c], 80 channels.
// m1 is [c][i][k] (rows K-contiguous -> A-frags), m2T is [c][j][i]
// (rows K-contiguous -> B-frags). Wave tile = 16 x 64 (1 A-frag x 4 B-frags
// -> 4 WMMAs); block = 64 x 64. Output written channel-last into skipin.
// ---------------------------------------------------------------------------
__global__ __launch_bounds__(128)
void chanmm_wmma(const __bf16* __restrict__ A, const __bf16* __restrict__ BT,
                 __bf16* __restrict__ Y, int xoff) {
  const int tid  = threadIdx.x;
  const int wave = tid >> 5, lane = tid & 31;
  const int half = lane >> 4, lr = lane & 15;
  const int c  = blockIdx.z;
  const int m0 = blockIdx.y * 64 + wave * 16;
  const int n0 = blockIdx.x * 64;
  const __bf16* Ar = A  + (size_t)c * NNSQ + (size_t)(m0 + lr) * NDIM;
  const __bf16* Br = BT + (size_t)c * NNSQ + (size_t)(n0 + lr) * NDIM;

  v8f acc[4] = {};
  for (int kk = 0; kk < NDIM; kk += 32) {
    v16bf a = mk16(*(const v8bf*)(Ar + kk + (half ? 16 : 0)),
                   *(const v8bf*)(Ar + kk + 8 + (half ? 16 : 0)));
    if (kk + 32 < NDIM) {   // global_prefetch_b8 for the next K-tile
      __builtin_prefetch(Ar + kk + 32, 0, 1);
      __builtin_prefetch(Br + kk + 32, 0, 1);
    }
#pragma unroll
    for (int t = 0; t < 4; ++t) {
      const __bf16* bp = Br + (size_t)t * 16 * NDIM + kk + (half ? 16 : 0);
      v16bf b = mk16(*(const v8bf*)bp, *(const v8bf*)(bp + 8));
      acc[t] = __builtin_amdgcn_wmma_f32_16x16x32_bf16(false, a, false, b,
                                                       (short)0, acc[t],
                                                       false, false);
    }
  }

#pragma unroll
  for (int t = 0; t < 4; ++t) {
    int j = n0 + t * 16 + lr;
#pragma unroll
    for (int r = 0; r < 8; ++r) {
      int i = m0 + half * 8 + r;
      Y[(size_t)(i * NDIM + j) * 160 + xoff + c] = f2bf(acc[t][r]);
    }
  }
}

// ---------------------------------------------------------------------------
// Small helper kernels
// ---------------------------------------------------------------------------
// raw input [17][NN] fp32 -> channel-last bf16 [NN][32]
__global__ void convert_x(const float* __restrict__ X, __bf16* __restrict__ Y) {
  int idx = blockIdx.x * blockDim.x + threadIdx.x;
  if (idx >= 17 * NNSQ) return;
  int c = idx / NNSQ, pos = idx - c * NNSQ;
  Y[(size_t)pos * 32 + c] = f2bf(X[idx]);
}

// channel-last copy: Y[pos][0:C] = X[pos][0:C]
__global__ void copy_cl(const __bf16* __restrict__ X, int ldx,
                        __bf16* __restrict__ Y, int ldy, int C) {
  int idx = blockIdx.x * blockDim.x + threadIdx.x;
  if (idx >= NNSQ * C) return;
  int pos = idx / C, c = idx - pos * C;
  Y[(size_t)pos * ldy + c] = X[(size_t)pos * ldx + c];
}

// nf[j][coff+c] = sum_i X[c][i][j]  (fp32 channel-first source: raw input)
__global__ void colsum_f32(const float* __restrict__ X, float* __restrict__ nf,
                           int C, int coff) {
  int idx = blockIdx.x * blockDim.x + threadIdx.x;
  if (idx >= C * NDIM) return;
  int c = idx / NDIM, j = idx - c * NDIM;
  float s = 0.f;
  for (int i = 0; i < NDIM; ++i) s += X[(size_t)c * NNSQ + i * NDIM + j];
  nf[j * CTOT + coff + c] = s;
}

// nf[j][coff+c] = sum_i X[(i*384+j)][c]  (channel-last bf16 source)
__global__ void colsum_cl(const __bf16* __restrict__ X, float* __restrict__ nf,
                          int ld, int C, int coff) {
  int idx = blockIdx.x * blockDim.x + threadIdx.x;
  if (idx >= NDIM * C) return;
  int j = idx / C, c = idx - j * C;
  float s = 0.f;
  for (int i = 0; i < NDIM; ++i)
    s += bf2f(X[(size_t)(i * NDIM + j) * ld + c]);
  nf[j * CTOT + coff + c] = s;
}

// ebf[e][0:257] = nf[srcs[e]], ebf[e][257:514] = nf[dsts[e]]  (ld = 544)
__global__ void edge_build(const float* __restrict__ nf,
                           const int* __restrict__ srcs,
                           const int* __restrict__ dsts,
                           __bf16* __restrict__ ebf) {
  int idx = blockIdx.x * blockDim.x + threadIdx.x;   // over E * 514
  if (idx >= EEDGE * 514) return;
  int e = idx / 514, col = idx - e * 514;
  int node = (col < CTOT) ? srcs[e] : dsts[e];
  int cc   = (col < CTOT) ? col : col - CTOT;
  ebf[(size_t)e * 544 + col] = f2bf(nf[node * CTOT + cc]);
}

// final 128 -> 2 linear layer (tiny; VALU)
__global__ void readout_final(const __bf16* __restrict__ Y2,
                              const float* __restrict__ W,
                              const float* __restrict__ Bv,
                              float* __restrict__ out) {
  int e = blockIdx.x * blockDim.x + threadIdx.x;
  if (e >= EEDGE) return;
  float a0 = Bv[0], a1 = Bv[1];
  for (int k = 0; k < 128; ++k) {
    float v = bf2f(Y2[(size_t)e * 128 + k]);
    a0 += v * W[k];
    a1 += v * W[128 + k];
  }
  out[e * 2 + 0] = a0;
  out[e * 2 + 1] = a1;
}

// ---------------------------------------------------------------------------
extern "C" void kernel_launch(void* const* d_in, const int* in_sizes, int n_in,
                              void* d_out, int out_size, void* d_ws, size_t ws_size,
                              hipStream_t stream) {
  (void)in_sizes; (void)n_in; (void)out_size; (void)ws_size;
  const float* x0   = (const float*)d_in[0];
  const int*   srcs = (const int*)d_in[2];
  const int*   dsts = (const int*)d_in[3];
  auto P = [&](int i) { return (const float*)d_in[i]; };

  // workspace layout
  char*  ws  = (char*)d_ws;
  size_t off = 0;
  auto alloc = [&](size_t bytes) -> void* {
    void* p = ws + off;
    off += (bytes + 255) & ~(size_t)255;
    return p;
  };
  float*  nf     = (float*) alloc((size_t)NDIM * CTOT * 4);
  __bf16* xb0    = (__bf16*)alloc((size_t)NNSQ * 32 * 2);    // [pos][32] (17 real)
  __bf16* tmp    = (__bf16*)alloc((size_t)NNSQ * 96 * 2);    // [pos][96] (80 real)
  __bf16* hcur   = (__bf16*)alloc((size_t)NNSQ * 96 * 2);    // [pos][96] (80 real)
  size_t ro_off  = off;                                      // readout aliases here
  __bf16* m1     = (__bf16*)alloc((size_t)80 * NNSQ * 2);    // [c][i][k]
  __bf16* m2T    = (__bf16*)alloc((size_t)80 * NNSQ * 2);    // [c][j][i]
  __bf16* skipin = (__bf16*)alloc((size_t)NNSQ * 160 * 2);   // [pos][160]
  // readout buffers alias the (dead after blocks) m1/m2T/skipin region
  size_t ro = ro_off;
  __bf16* ebf = (__bf16*)(ws + ro);  ro += (size_t)EEDGE * 544 * 2;
  __bf16* y1  = (__bf16*)(ws + ro);  ro += (size_t)EEDGE * 288 * 2;
  __bf16* y2  = (__bf16*)(ws + ro);

  // zero channel-last buffers so interleaved pad channels stay zero
  hipMemsetAsync(xb0,    0, (size_t)NNSQ * 32  * 2, stream);
  hipMemsetAsync(tmp,    0, (size_t)NNSQ * 96  * 2, stream);
  hipMemsetAsync(hcur,   0, (size_t)NNSQ * 96  * 2, stream);
  hipMemsetAsync(skipin, 0, (size_t)NNSQ * 160 * 2, stream);

  // x -> channel-last bf16; x0 column sums -> node features [j][0:17]
  convert_x<<<(17 * NNSQ + 255) / 256, 256, 0, stream>>>(x0, xb0);
  colsum_f32<<<(17 * NDIM + 255) / 256, 256, 0, stream>>>(x0, nf, 17, 0);

  const dim3 cg(NNSQ / 64, 5);   // conv grid: 2304 row tiles x 5 out-ch tiles
  const dim3 mg(6, 6, 80);       // matmul grid: 64x64 tiles x 80 channels

  for (int b = 0; b < 3; ++b) {
    int wb  = 4 + 10 * b;
    const __bf16* xin = (b == 0) ? xb0 : hcur;
    int Kin = (b == 0) ? 32 : 96;
    int Cin = (b == 0) ? 17 : 80;

    // mlp1 (channel-last hidden, then m1 channel-first for the matmul)
    gemm_xwT_wmma<<<cg, 128, 0, stream>>>(xin, P(wb + 0), P(wb + 1), tmp,
                                          Kin, Cin, 80, 96, 1, 0);
    gemm_xwT_wmma<<<cg, 128, 0, stream>>>(tmp, P(wb + 2), P(wb + 3), m1,
                                          96, 80, 80, 0, 1, 1);
    // mlp2 (m2 written channel-first TRANSPOSED for direct B-fragments)
    gemm_xwT_wmma<<<cg, 128, 0, stream>>>(xin, P(wb + 4), P(wb + 5), tmp,
                                          Kin, Cin, 80, 96, 1, 0);
    gemm_xwT_wmma<<<cg, 128, 0, stream>>>(tmp, P(wb + 6), P(wb + 7), m2T,
                                          96, 80, 80, 0, 1, 2);

    // skipin[pos][0:Cin] = x ; skipin[pos][Cin:Cin+80] = m1 @ m2
    copy_cl<<<((size_t)NNSQ * Cin + 255) / 256, 256, 0, stream>>>(
        xin, Kin, skipin, 160, Cin);
    chanmm_wmma<<<mg, 128, 0, stream>>>(m1, m2T, skipin, Cin);

    // skip conv (no relu) -> new x (channel-last)
    int Csk = (b == 0) ? 97 : 160;
    gemm_xwT_wmma<<<cg, 128, 0, stream>>>(skipin, P(wb + 8), P(wb + 9), hcur,
                                          160, Csk, 80, 96, 0, 0);
    // node-feature column sums for this block's output
    colsum_cl<<<(NDIM * 80 + 255) / 256, 256, 0, stream>>>(hcur, nf, 96, 80,
                                                           17 + 80 * b);
  }

  // readout: gather per-edge features, two WMMA GEMMs, tiny final layer
  hipMemsetAsync(ebf, 0, (size_t)EEDGE * 544 * 2, stream);
  hipMemsetAsync(y1,  0, (size_t)EEDGE * 288 * 2, stream);
  edge_build<<<(EEDGE * 514 + 255) / 256, 256, 0, stream>>>(nf, srcs, dsts, ebf);
  gemm_xwT_wmma<<<dim3(EEDGE / 64, 17), 128, 0, stream>>>(ebf, P(34), P(35), y1,
                                                          544, 514, 257, 288, 1, 0);
  gemm_xwT_wmma<<<dim3(EEDGE / 64, 8), 128, 0, stream>>>(y1, P(36), P(37), y2,
                                                         288, 257, 128, 128, 1, 0);
  readout_final<<<(EEDGE + 255) / 256, 256, 0, stream>>>(y2, P(38), P(39),
                                                         (float*)d_out);
}
